// QuantLinear_8358006358511
// MI455X (gfx1250) — matrix operand order repaired and checked by reading, not compiled, not run on hardware
//
#include <hip/hip_runtime.h>

typedef __attribute__((ext_vector_type(2)))  _Float16 h2;
typedef __attribute__((ext_vector_type(8)))  _Float16 v8h;
typedef __attribute__((ext_vector_type(16))) _Float16 v16h;
typedef __attribute__((ext_vector_type(8)))  float    v8f;
typedef __attribute__((ext_vector_type(4)))  int      v4i;

namespace {
constexpr int K  = 4096;
constexpr int N  = 4096;
constexpr int GS = 128;              // quant group size
constexpr int G  = K / GS;           // 32 groups
constexpr int BM = 128;
constexpr int BN = 256;              // widened: 64x64 wave tiles
constexpr int BK = 64;               // divides GS -> one (zero,scale) per staged row
constexpr int NT = K / BK;           // 64 k-tiles
constexpr int LDA = BK + 8;          // halves, padded vs bank conflicts
constexpr int LDB = BK + 8;
constexpr int SA_STAGE = BM * LDA;   // 9216 halves per stage
constexpr int SB_STAGE = BN * LDB;   // 18432 halves per stage
constexpr unsigned SMEM_BYTES =
    (unsigned)(2 * SA_STAGE + 2 * SB_STAGE) * (unsigned)sizeof(_Float16); // 110592
}

static __device__ __forceinline__ v16h cat8(v8h lo, v8h hi) {
  return __builtin_shufflevector(lo, hi, 0,1,2,3,4,5,6,7,8,9,10,11,12,13,14,15);
}

__global__ __launch_bounds__(256)
void w4a16_wmma_gemm(const _Float16* __restrict__ A,
                     const int*      __restrict__ qw,
                     const int*      __restrict__ zr,
                     const _Float16* __restrict__ sc,
                     const _Float16* __restrict__ bias,
                     float*          __restrict__ C)
{
  extern __shared__ _Float16 smem[];
  _Float16* sA = smem;                    // [2][BM*LDA]
  _Float16* sB = smem + 2 * SA_STAGE;     // [2][BN*LDB]

  const int tid  = threadIdx.x;
  const int lane = tid & 31;
  const int wave = tid >> 5;              // 0..7
  const int wm   = (wave >> 2) * 64;      // 2 waves tile M (64 rows each)
  const int wn   = (wave & 3) * 64;       // 4 waves tile N (64 cols each)
  const int lm   = lane & 15;
  const int kh   = lane >> 4;

  const int bm = blockIdx.y * BM;
  const int bn = blockIdx.x * BN;

  const int row  = tid >> 1;              // A staging row 0..127
  const int half = tid & 1;
  const int n    = bn + tid;              // B staging row 0..255 (one per thread)

  const _Float16* aSrcBase = A  + (size_t)(bm + row) * K + half * 32;
  const int*      qSrcBase = qw + (size_t)n * (K / 2);

  v8f acc[4][4] = {};
  v8h areg[4];                            // prefetched A (32 halves)
  v4i qreg[8];                            // prefetched packed weights (32 ints = 1 row)
  int   zq    = 0;
  float scale = 0.f;

  // ---- fetch tile kt's globals into registers (latency hidden by compute) ----
  auto fetch = [&](int kt) {
    const _Float16* as = aSrcBase + kt * BK;
    const int*      qs = qSrcBase + kt * (BK / 2);
#pragma unroll
    for (int i = 0; i < 4; ++i) areg[i] = *(const v8h*)(as + i * 8);
#pragma unroll
    for (int i = 0; i < 8; ++i) qreg[i] = *(const v4i*)(qs + i * 4);
    const int g = (kt * BK) / GS;
    zq    = zr[g * (N / 2) + (n >> 1)];
    scale = (float)sc[n * G + g];
    if (kt + 1 < NT) {                    // warm caches for the tile after next
      __builtin_prefetch(as + BK, 0, 1);
      __builtin_prefetch(qs + BK / 2, 0, 1);
      __builtin_prefetch(qs + BK / 2 + 16, 0, 1);
    }
  };

  // ---- dequant + write registers into LDS stage s ----
  // fp16 magic: (0x6400 | nib) == 1024 + nib exactly; one v_pk_fma_f16 then
  // computes (nib - zp) * scale for a nibble pair.
  auto stage = [&](int s) {
    _Float16* ad = sA + s * SA_STAGE + row * LDA + half * 32;
#pragma unroll
    for (int i = 0; i < 4; ++i) *(v8h*)(ad + i * 8) = areg[i];

    const int   zpv = (zq >> ((n & 1) * 4)) & 15;
    const _Float16 sh = (_Float16)scale;
    const _Float16 bh = (_Float16)(-(1024.0f + (float)zpv) * scale);
    const h2 s2 = {sh, sh};
    const h2 z2 = {bh, bh};

    _Float16* bd = sB + s * SB_STAGE + tid * LDB;
#pragma unroll
    for (int i = 0; i < 8; ++i) {
      const v4i q = qreg[i];
      v8h o;
#pragma unroll
      for (int j = 0; j < 4; ++j) {
        const unsigned b = (unsigned)q[j];
        const unsigned t = ((b | (b << 12)) & 0x000F000Fu) | 0x64006400u;
        const h2 x = __builtin_bit_cast(h2, t);
        const h2 r = x * s2 + z2;         // v_pk_fma_f16
        o[2 * j]     = r[0];
        o[2 * j + 1] = r[1];
      }
      *(v8h*)(bd + i * 8) = o;
    }
  };

  // ---- 32 WMMAs on LDS stage s (1.0 ds_load per WMMA) ----
  auto compute = [&](int s) {
    const _Float16* aBase = sA + s * SA_STAGE;
    const _Float16* bBase = sB + s * SB_STAGE;
#pragma unroll
    for (int ks = 0; ks < BK / 32; ++ks) {
      const int kk = ks * 32;
      v16h bfrag[4];
#pragma unroll
      for (int nf = 0; nf < 4; ++nf) {
        const _Float16* p = bBase + (wn + nf * 16 + lm) * LDB + kk + kh * 16;
        bfrag[nf] = cat8(*(const v8h*)p, *(const v8h*)(p + 8));
      }
#pragma unroll
      for (int mf = 0; mf < 4; ++mf) {
        const _Float16* p = aBase + (wm + mf * 16 + lm) * LDA + kk + kh * 16;
        const v16h afrag = cat8(*(const v8h*)p, *(const v8h*)(p + 8));
#pragma unroll
        for (int nf = 0; nf < 4; ++nf) {
          acc[mf][nf] = __builtin_amdgcn_wmma_f32_16x16x32_f16(
              false, afrag, false, bfrag[nf], (short)0, acc[mf][nf], false, false);
        }
      }
    }
  };

  // ---- software pipeline: fetch k+1 / compute k / stage k+1 ----
  fetch(0);
  stage(0);
  __syncthreads();

  for (int kt = 0; kt < NT - 1; ++kt) {
    const int s = kt & 1;
    fetch(kt + 1);
    compute(s);
    stage(s ^ 1);
    __syncthreads();
  }
  compute((NT - 1) & 1);

  // ---- epilogue: D layout VGPR r -> M = r + 8*kh, N = lm ----
#pragma unroll
  for (int nf = 0; nf < 4; ++nf) {
    const int col = bn + wn + nf * 16 + lm;
    const float bv = (float)bias[col];
#pragma unroll
    for (int mf = 0; mf < 4; ++mf) {
#pragma unroll
      for (int r = 0; r < 8; ++r) {
        const int rowg = bm + wm + mf * 16 + kh * 8 + r;
        C[(size_t)rowg * N + col] = acc[mf][nf][r] + bv;
      }
    }
  }
}

extern "C" void kernel_launch(void* const* d_in, const int* in_sizes, int n_in,
                              void* d_out, int out_size, void* d_ws, size_t ws_size,
                              hipStream_t stream) {
  const _Float16* A    = (const _Float16*)d_in[0];
  const int*      qw   = (const int*)d_in[1];
  const int*      zr   = (const int*)d_in[2];
  const _Float16* sc   = (const _Float16*)d_in[3];
  const _Float16* bias = (const _Float16*)d_in[4];
  float* C = (float*)d_out;

  const int M = in_sizes[0] / K;        // A is [M, K] fp16
  dim3 grid(N / BN, M / BM);
  w4a16_wmma_gemm<<<grid, dim3(256, 1, 1), SMEM_BYTES, stream>>>(A, qw, zr, sc, bias, C);
}
